// PointMLPEncoder_19387482374221
// MI455X (gfx1250) — matrix operand order repaired and checked by reading, not compiled, not run on hardware
//
#include <hip/hip_runtime.h>
#include <hip/hip_bf16.h>
#include <math.h>

#define KNN_K 24

typedef __attribute__((ext_vector_type(8)))  float  v8f;
typedef __attribute__((ext_vector_type(8)))  __bf16 v8bf;
typedef __attribute__((ext_vector_type(16))) __bf16 v16bf;
typedef __attribute__((ext_vector_type(4)))  int    v4i;

#if defined(__HIP_DEVICE_COMPILE__) && defined(__has_builtin)
#if __has_builtin(__builtin_amdgcn_global_load_async_to_lds_b128) && \
    __has_builtin(__builtin_amdgcn_s_wait_asynccnt)
#define PM_ASYNC_LDS 1
#endif
#endif

#ifdef PM_ASYNC_LDS
typedef __attribute__((address_space(1))) v4i gv4i_t;   // global (AS1)
typedef __attribute__((address_space(3))) v4i lv4i_t;   // LDS (AS3)
__device__ __forceinline__ void pm_async_cp16(const __bf16* g, __bf16* l) {
  __builtin_amdgcn_global_load_async_to_lds_b128((gv4i_t*)g, (lv4i_t*)l, 0, 0);
}
#endif

// ---------------------------------------------------------------------------
// WMMA GEMM: C[M,N] = A[M,K] * Wt^T, A bf16 [M,K] row-major, Wt bf16 [N,K]
// (pre-transposed weights, k-contiguous).  f32 accumulate, f32 C output.
// M%64==0, N%128==0, K%32==0 hold for every layer.
// Block = 256 threads = 8 waves; block tile 64x128; wave tile 32x32 (2x2 WMMA).
// Panels staged global->LDS with GLOBAL_LOAD_ASYNC_TO_LDS_B128 when available.
// ---------------------------------------------------------------------------
#define LDS_STRIDE 40   // padded row length in bf16 elems (80 bytes, 16B mult)

__global__ __launch_bounds__(256)
void pm_gemm_bf16(const __bf16* __restrict__ A, const __bf16* __restrict__ Wt,
                  float* __restrict__ C, int M, int N, int K)
{
  __shared__ __attribute__((aligned(16))) __bf16 As[64  * LDS_STRIDE];
  __shared__ __attribute__((aligned(16))) __bf16 Bs[128 * LDS_STRIDE];

  const int m0   = blockIdx.x * 64;
  const int n0   = blockIdx.y * 128;
  const int tid  = threadIdx.x;
  const int lane = tid & 31;
  const int w    = tid >> 5;
  const int wm   = w & 1;        // 2 waves along M (32 rows each)
  const int wn   = w >> 1;       // 4 waves along N (32 cols each)
  const int half = lane >> 4;
  const int l16  = lane & 15;

  // staging coordinates: A = 256 16B-chunks, B = 512 16B-chunks
  const int ar  = tid >> 2;            // A row 0..63
  const int as_ = (tid & 3) * 8;       // A k-seg {0,8,16,24}
  const int br0 = tid >> 2;            // B rows for chunk pair
  const int br1 = (tid + 256) >> 2;
  const int bs_ = (tid & 3) * 8;

  v8f acc[2][2] = {};

  for (int kk = 0; kk < K; kk += 32) {
    const __bf16* ga  = A  + (size_t)(m0 + ar)  * K + kk + as_;
    const __bf16* gb0 = Wt + (size_t)(n0 + br0) * K + kk + bs_;
    const __bf16* gb1 = Wt + (size_t)(n0 + br1) * K + kk + bs_;
#ifdef PM_ASYNC_LDS
    pm_async_cp16(ga,  &As[ar  * LDS_STRIDE + as_]);
    pm_async_cp16(gb0, &Bs[br0 * LDS_STRIDE + bs_]);
    pm_async_cp16(gb1, &Bs[br1 * LDS_STRIDE + bs_]);
    __builtin_amdgcn_s_wait_asynccnt(0);
#else
    *(v8bf*)&As[ar  * LDS_STRIDE + as_] = *(const v8bf*)ga;
    *(v8bf*)&Bs[br0 * LDS_STRIDE + bs_] = *(const v8bf*)gb0;
    *(v8bf*)&Bs[br1 * LDS_STRIDE + bs_] = *(const v8bf*)gb1;
#endif
    if (kk + 32 < K) __builtin_prefetch(ga + 32, 0, 0);
    __syncthreads();

    // ---- fragments (layouts per CDNA5 ISA 7.12.2) ----
    v16bf af[2], bfr[2];
#pragma unroll
    for (int ms = 0; ms < 2; ++ms) {
      const __bf16* p = &As[(wm * 32 + ms * 16 + l16) * LDS_STRIDE];
      v8bf lo = *(const v8bf*)(p + half * 8);        // K = h*8 .. h*8+7
      v8bf hi = *(const v8bf*)(p + 16 + half * 8);   // K = 16+h*8 ..
      af[ms] = __builtin_shufflevector(lo, hi, 0,1,2,3,4,5,6,7,8,9,10,11,12,13,14,15);
    }
#pragma unroll
    for (int ns = 0; ns < 2; ++ns) {
      const __bf16* p = &Bs[(wn * 32 + ns * 16 + l16) * LDS_STRIDE];
      v8bf lo = *(const v8bf*)(p + half * 8);        // K = h*8 .. h*8+7
      v8bf hi = *(const v8bf*)(p + 16 + half * 8);   // K = 16+h*8 ..
      bfr[ns] = __builtin_shufflevector(lo, hi, 0,1,2,3,4,5,6,7,8,9,10,11,12,13,14,15);
    }
#pragma unroll
    for (int ms = 0; ms < 2; ++ms)
#pragma unroll
      for (int ns = 0; ns < 2; ++ns)
        acc[ms][ns] = __builtin_amdgcn_wmma_f32_16x16x32_bf16(
            false, af[ms], false, bfr[ns], (short)0, acc[ms][ns], false, false);
    __syncthreads();
  }

  // ---- store: lane(l16)=N, vgpr v -> M = v + 8*half ----
#pragma unroll
  for (int ms = 0; ms < 2; ++ms)
#pragma unroll
    for (int ns = 0; ns < 2; ++ns) {
      int n = n0 + wn * 32 + ns * 16 + l16;
#pragma unroll
      for (int v = 0; v < 8; ++v) {
        int m = m0 + wm * 32 + ms * 16 + v + 8 * half;
        C[(size_t)m * N + n] = acc[ms][ns][v];
      }
    }
}

// ---------------------------------------------------------------------------
// weight convert + transpose: W[K,N] f32 -> Wt[N,K] bf16 (once per launch)
// ---------------------------------------------------------------------------
__global__ void pm_wcvt(const float* __restrict__ W, __bf16* __restrict__ Wt,
                        int K, int N)
{
  long i = (long)blockIdx.x * blockDim.x + threadIdx.x;
  if (i >= (long)K * N) return;
  int k = (int)(i / N), n = (int)(i % N);
  Wt[(size_t)n * K + k] = (__bf16)W[i];
}

// ---------------------------------------------------------------------------
// init embed: out[r,c] = x[r]*W[c] + b[c]   (K=1), bf16 activations out
// ---------------------------------------------------------------------------
__global__ void pm_embed(const float* __restrict__ x, const float* __restrict__ W,
                         const float* __restrict__ b, __bf16* __restrict__ out,
                         long total, int d)
{
  long i = (long)blockIdx.x * blockDim.x + threadIdx.x;
  if (i >= total) return;
  int c = (int)(i % d);
  out[i] = (__bf16)(x[i / d] * W[c] + b[c]);
}

// ---------------------------------------------------------------------------
// farthest-point sampling: one block per batch, deterministic LDS argmax
// ---------------------------------------------------------------------------
__global__ __launch_bounds__(256)
void pm_fps(const float* __restrict__ pos, int* __restrict__ idx_out, int n, int m)
{
  const int tid = threadIdx.x;
  const float* pb = pos + (long)blockIdx.x * n * 3;
  int* ib = idx_out + (long)blockIdx.x * m;
  __shared__ float sd[256];
  __shared__ int   si[256];
  __shared__ int   scur;
  float mind[32];
  const int npt = (n + 255) >> 8;
  for (int j = 0; j < npt; ++j) mind[j] = 3.4e38f;
  if (tid == 0) scur = 0;
  __syncthreads();
  for (int it = 0; it < m; ++it) {
    int cur = scur;
    if (tid == 0) ib[it] = cur;
    float cx = pb[cur*3+0], cy = pb[cur*3+1], cz = pb[cur*3+2];
    float best = -1.0f; int bi = 0;
    for (int j = 0; j < npt; ++j) {
      int p = (j << 8) + tid;
      if (p < n) {
        float dx = pb[p*3+0]-cx, dy = pb[p*3+1]-cy, dz = pb[p*3+2]-cz;
        float d = dx*dx + dy*dy + dz*dz;
        float md = fminf(mind[j], d);
        mind[j] = md;
        if (md > best) { best = md; bi = p; }
      }
    }
    sd[tid] = best; si[tid] = bi;
    __syncthreads();
    for (int s = 128; s > 0; s >>= 1) {
      if (tid < s && sd[tid+s] > sd[tid]) { sd[tid] = sd[tid+s]; si[tid] = si[tid+s]; }
      __syncthreads();
    }
    if (tid == 0) scur = si[0];
    __syncthreads();
  }
}

// ---------------------------------------------------------------------------
// kNN: one thread per anchor, 24-entry insertion list with threshold skip
// ---------------------------------------------------------------------------
__global__ void pm_knn(const float* __restrict__ pos, const int* __restrict__ idx,
                       int* __restrict__ nbr, int n, int m)
{
  int t = blockIdx.x * blockDim.x + threadIdx.x;
  if (t >= 4 * m) return;
  int b = t / m;
  const float* pb = pos + (long)b * n * 3;
  int ai = idx[t];
  float ax = pb[ai*3+0], ay = pb[ai*3+1], az = pb[ai*3+2];
  float dist[KNN_K]; int ind[KNN_K];
#pragma unroll
  for (int j = 0; j < KNN_K; ++j) { dist[j] = 3.4e38f; ind[j] = 0; }
  for (int p = 0; p < n; ++p) {
    float dx = pb[p*3+0]-ax, dy = pb[p*3+1]-ay, dz = pb[p*3+2]-az;
    float d = dx*dx + dy*dy + dz*dz;
    if (d < dist[KNN_K-1]) {
      int j = KNN_K - 1;
      while (j > 0 && dist[j-1] > d) { dist[j] = dist[j-1]; ind[j] = ind[j-1]; --j; }
      dist[j] = d; ind[j] = p;
    }
  }
#pragma unroll
  for (int j = 0; j < KNN_K; ++j) nbr[(long)t * KNN_K + j] = ind[j];
}

__global__ void pm_gather_pos(const float* __restrict__ pos, const int* __restrict__ idx,
                              float* __restrict__ out, int n, int m, long total)
{
  long i = (long)blockIdx.x * blockDim.x + threadIdx.x;
  if (i >= total) return;
  int j = (int)(i % 3);
  long r = i / 3;
  int a = (int)(r % m), b = (int)(r / m);
  out[i] = pos[((long)b * n + idx[b * m + a]) * 3 + j];
}

// ---------------------------------------------------------------------------
// sigma = sqrt(max(sum((x_nb-x_an)^2)/(K*n_tot*d), 1e-6)) — deterministic 2-pass
// ---------------------------------------------------------------------------
__global__ __launch_bounds__(256)
void pm_sigma_partial(const __bf16* __restrict__ x, const int* __restrict__ idx,
                      const int* __restrict__ nbr, float* __restrict__ partial,
                      int n, int m, int d, long total)
{
  float acc = 0.0f;
  for (long i = (long)blockIdx.x * blockDim.x + threadIdx.x; i < total;
       i += (long)gridDim.x * blockDim.x) {
    int c = (int)(i % d);
    long r = i / d;
    int kk = (int)(r % KNN_K);
    long r2 = r / KNN_K;
    int a = (int)(r2 % m), b = (int)(r2 / m);
    int pn = nbr[((long)(b * m + a)) * KNN_K + kk];
    int an = idx[b * m + a];
    float diff = (float)x[((long)b * n + pn) * d + c] - (float)x[((long)b * n + an) * d + c];
    acc += diff * diff;
  }
  __shared__ float s[256];
  s[threadIdx.x] = acc; __syncthreads();
  for (int st = 128; st > 0; st >>= 1) {
    if (threadIdx.x < st) s[threadIdx.x] += s[threadIdx.x + st];
    __syncthreads();
  }
  if (threadIdx.x == 0) partial[blockIdx.x] = s[0];
}

__global__ __launch_bounds__(256)
void pm_reduce1(const float* __restrict__ partial, int nblk, float* __restrict__ out)
{
  float a = 0.0f;
  for (int i = threadIdx.x; i < nblk; i += 256) a += partial[i];
  __shared__ float s[256];
  s[threadIdx.x] = a; __syncthreads();
  for (int st = 128; st > 0; st >>= 1) {
    if (threadIdx.x < st) s[threadIdx.x] += s[threadIdx.x + st];
    __syncthreads();
  }
  if (threadIdx.x == 0) out[0] = s[0];
}

__global__ void pm_scale(__bf16* __restrict__ x, const float* __restrict__ alpha,
                         const float* __restrict__ beta, const float* __restrict__ sum,
                         float denom, long total, int d)
{
  long i = (long)blockIdx.x * blockDim.x + threadIdx.x;
  if (i >= total) return;
  float sigma = sqrtf(fmaxf(sum[0] * denom, 1e-6f));
  int c = (int)(i % d);
  x[i] = (__bf16)((float)x[i] * (alpha[c] / (sigma + 1e-5f)) + beta[c]);
}

// ---------------------------------------------------------------------------
// BatchNorm (training-mode) stats: deterministic two-pass column reduction
// ---------------------------------------------------------------------------
__global__ __launch_bounds__(256)
void pm_colstats_partial(const float* __restrict__ X, float* __restrict__ psum,
                         float* __restrict__ psq, int M, int N)
{
  int r0 = blockIdx.x * 256;
  int rend = r0 + 256; if (rend > M) rend = M;
  for (int c = threadIdx.x; c < N; c += 256) {
    float s = 0.0f, sq = 0.0f;
    for (int r = r0; r < rend; ++r) {
      float v = X[(size_t)r * N + c];
      s += v; sq += v * v;
    }
    psum[(size_t)blockIdx.x * N + c] = s;
    psq [(size_t)blockIdx.x * N + c] = sq;
  }
}

__global__ void pm_colstats_final(const float* __restrict__ psum, const float* __restrict__ psq,
                                  float* __restrict__ mean, float* __restrict__ rstd,
                                  int RB, int M, int N)
{
  int c = blockIdx.x * blockDim.x + threadIdx.x;
  if (c >= N) return;
  float s = 0.0f, sq = 0.0f;
  for (int rb = 0; rb < RB; ++rb) { s += psum[(size_t)rb * N + c]; sq += psq[(size_t)rb * N + c]; }
  float mu = s / (float)M;
  float var = sq / (float)M - mu * mu;
  mean[c] = mu;
  rstd[c] = rsqrtf(var + 1e-5f);
}

__global__ void pm_bn_relu(const float* __restrict__ X, const float* __restrict__ mean,
                           const float* __restrict__ rstd, const float* __restrict__ g,
                           const float* __restrict__ bb, __bf16* __restrict__ out,
                           long total, int N)
{
  long i = (long)blockIdx.x * blockDim.x + threadIdx.x;
  if (i >= total) return;
  int c = (int)(i % N);
  out[i] = (__bf16)fmaxf((X[i] - mean[c]) * rstd[c] * g[c] + bb[c], 0.0f);
}

__global__ void pm_bn_add(const float* __restrict__ X, const float* __restrict__ mean,
                          const float* __restrict__ rstd, const float* __restrict__ g,
                          const float* __restrict__ bb, const __bf16* __restrict__ res,
                          __bf16* __restrict__ out, long total, int N)
{
  long i = (long)blockIdx.x * blockDim.x + threadIdx.x;
  if (i >= total) return;
  int c = (int)(i % N);
  out[i] = (__bf16)((X[i] - mean[c]) * rstd[c] * g[c] + bb[c] + (float)res[i]);
}

// ---------------------------------------------------------------------------
// max aggregation over k neighbors, and global max pool
// ---------------------------------------------------------------------------
__global__ void pm_gather_max(const __bf16* __restrict__ h, const int* __restrict__ nbr,
                              __bf16* __restrict__ out, int m, int n, int d, long total)
{
  long i = (long)blockIdx.x * blockDim.x + threadIdx.x;
  if (i >= total) return;
  int c = (int)(i % d);
  long r = i / d;
  int a = (int)(r % m), b = (int)(r / m);
  const int* nb = nbr + ((long)(b * m + a)) * KNN_K;
  float mx = -3.4e38f;
#pragma unroll
  for (int k = 0; k < KNN_K; ++k)
    mx = fmaxf(mx, (float)h[((long)b * n + nb[k]) * d + c]);
  out[i] = (__bf16)mx;
}

__global__ void pm_global_max(const __bf16* __restrict__ h, float* __restrict__ g,
                              int m, int d)
{
  int i = blockIdx.x * blockDim.x + threadIdx.x;
  if (i >= 4 * d) return;
  int c = i % d, b = i / d;
  float mx = -3.4e38f;
  for (int r = 0; r < m; ++r) mx = fmaxf(mx, (float)h[((long)b * m + r) * d + c]);
  g[i] = mx;
}

// ---------------------------------------------------------------------------
// classifier layer: [4,Cin] @ [Cin,Cout] -> BN over batch of 4 -> ReLU
// (linear bias cancels exactly under training-mode BN)
// ---------------------------------------------------------------------------
__global__ void pm_classifier(const float* __restrict__ gin, const float* __restrict__ W,
                              const float* __restrict__ gbn, const float* __restrict__ bbn,
                              float* __restrict__ gout, int Cin, int Cout)
{
  int c = blockIdx.x * blockDim.x + threadIdx.x;
  if (c >= Cout) return;
  float y[4];
#pragma unroll
  for (int b = 0; b < 4; ++b) {
    float s = 0.0f;
    for (int k = 0; k < Cin; ++k) s += gin[b * Cin + k] * W[(size_t)k * Cout + c];
    y[b] = s;
  }
  float mu = (y[0] + y[1] + y[2] + y[3]) * 0.25f;
  float var = 0.0f;
#pragma unroll
  for (int b = 0; b < 4; ++b) { float t = y[b] - mu; var += t * t; }
  var *= 0.25f;
  float rs = rsqrtf(var + 1e-5f);
#pragma unroll
  for (int b = 0; b < 4; ++b)
    gout[b * Cout + c] = fmaxf((y[b] - mu) * rs * gbn[c] + bbn[c], 0.0f);
}

// ===========================================================================
// host launcher
// ===========================================================================
static inline unsigned cdiv(long a, long b) { return (unsigned)((a + b - 1) / b); }

extern "C" void kernel_launch(void* const* d_in, const int* in_sizes, int n_in,
                              void* d_out, int out_size, void* d_ws, size_t ws_size,
                              hipStream_t stream) {
  (void)in_sizes; (void)n_in; (void)out_size; (void)ws_size;

  // ---- unpack inputs (setup_inputs dict insertion order, recursively) ----
  int q = 0;
  const float* x    = (const float*)d_in[q++];
  const float* pos0 = (const float*)d_in[q++];
  const float* eW   = (const float*)d_in[q++];
  const float* eb   = (const float*)d_in[q++];
  struct ConvP {
    const float *alpha,*beta,*tW,*tb,*tg,*tbb;
    const float *pW1,*pg1,*pb1,*pW2,*pg2,*pb2;
    const float *qW1,*qg1,*qb1,*qW2,*qg2,*qb2;
  } cp[3];
  for (int i = 0; i < 3; ++i) {
    cp[i].alpha=(const float*)d_in[q++]; cp[i].beta=(const float*)d_in[q++];
    cp[i].tW=(const float*)d_in[q++]; cp[i].tb=(const float*)d_in[q++];
    cp[i].tg=(const float*)d_in[q++]; cp[i].tbb=(const float*)d_in[q++];
    cp[i].pW1=(const float*)d_in[q++]; cp[i].pg1=(const float*)d_in[q++]; cp[i].pb1=(const float*)d_in[q++];
    cp[i].pW2=(const float*)d_in[q++]; cp[i].pg2=(const float*)d_in[q++]; cp[i].pb2=(const float*)d_in[q++];
    cp[i].qW1=(const float*)d_in[q++]; cp[i].qg1=(const float*)d_in[q++]; cp[i].qb1=(const float*)d_in[q++];
    cp[i].qW2=(const float*)d_in[q++]; cp[i].qg2=(const float*)d_in[q++]; cp[i].qb2=(const float*)d_in[q++];
  }
  const float* cW[2]; const float* cg[2]; const float* cbb[2];
  for (int i = 0; i < 2; ++i) {
    cW[i]=(const float*)d_in[q++]; q++; /* linear bias cancels under BN */
    cg[i]=(const float*)d_in[q++]; cbb[i]=(const float*)d_in[q++];
  }

  // ---- workspace bump allocator ----
  char* wsp = (char*)d_ws;
  auto alloc = [&](size_t bytes) -> void* {
    void* r = (void*)wsp; wsp += (bytes + 255) & ~(size_t)255; return r;
  };
  __bf16* X0  = (__bf16*)alloc((size_t)4*8192*64 *2);   // embed out / ping (bf16)
  __bf16* X1  = (__bf16*)alloc((size_t)4*2048*128*2);   // pong (all conv outs fit)
  __bf16* ACTA= (__bf16*)alloc((size_t)4*8192*128*2);   // activation buffers
  __bf16* ACTB= (__bf16*)alloc((size_t)4*8192*128*2);
  float*  CB  = (float*) alloc((size_t)4*8192*128*4);   // f32 GEMM output
  __bf16* AGG = (__bf16*)alloc((size_t)4*2048*128*2);
  float* PA  = (float*)alloc((size_t)4*2048*3*4);
  float* PB  = (float*)alloc((size_t)4*2048*3*4);
  int*   IDX = (int*)  alloc((size_t)4*2048*4);
  int*   NBR = (int*)  alloc((size_t)4*2048*KNN_K*4);
  float* PSUM= (float*)alloc((size_t)128*512*4);
  float* PSQ = (float*)alloc((size_t)128*512*4);
  float* MEAN= (float*)alloc(512*4);
  float* RSTD= (float*)alloc(512*4);
  float* SIGP= (float*)alloc(1024*4);
  float* SIG = (float*)alloc(256);
  float* GP  = (float*)alloc((size_t)4*512*4);
  float* CT  = (float*)alloc((size_t)4*256*4);

  // ---- bf16 transposed weight copies (once per launch) ----
  auto wcvt = [&](const float* W, int K, int N) -> __bf16* {
    __bf16* Wt = (__bf16*)alloc((size_t)K * N * 2);
    pm_wcvt<<<cdiv((long)K * N, 256), 256, 0, stream>>>(W, Wt, K, N);
    return Wt;
  };
  const int diArr[3] = {64, 128, 256};
  const int doArr[3] = {128, 256, 512};
  __bf16 *wtT[3], *wtP1[3], *wtP2[3], *wtQ1[3], *wtQ2[3];
  for (int i = 0; i < 3; ++i) {
    wtT [i] = wcvt(cp[i].tW,  diArr[i], doArr[i]);
    wtP1[i] = wcvt(cp[i].pW1, doArr[i], doArr[i]);
    wtP2[i] = wcvt(cp[i].pW2, doArr[i], doArr[i]);
    wtQ1[i] = wcvt(cp[i].qW1, doArr[i], doArr[i]);
    wtQ2[i] = wcvt(cp[i].qW2, doArr[i], doArr[i]);
  }

  auto gemm = [&](const __bf16* A, const __bf16* Wt, float* Cmat, int M, int N, int K) {
    dim3 grid((unsigned)(M / 64), (unsigned)(N / 128));
    pm_gemm_bf16<<<grid, 256, 0, stream>>>(A, Wt, Cmat, M, N, K);
  };
  auto bnstats = [&](const float* Xp, int M, int N) {
    int RB = (M + 255) / 256;
    pm_colstats_partial<<<RB, 256, 0, stream>>>(Xp, PSUM, PSQ, M, N);
    pm_colstats_final<<<cdiv(N,256), 256, 0, stream>>>(PSUM, PSQ, MEAN, RSTD, RB, M, N);
  };

  // ---- init embed: [4*8192,1] -> [4*8192,64] bf16 ----
  {
    long tot = (long)4*8192*64;
    pm_embed<<<cdiv(tot,256), 256, 0, stream>>>(x, eW, eb, X0, tot, 64);
  }

  const int nArr[3]  = {8192, 2048, 1024};
  const int mArr[3]  = {2048, 1024, 512};

  __bf16* xin = X0;
  const float* posin = pos0;
  for (int i = 0; i < 3; ++i) {
    const int n = nArr[i], m = mArr[i], di = diArr[i], dk = doArr[i];
    __bf16* xout  = (i & 1) ? X0 : X1;
    float* posout = (i & 1) ? PB : PA;

    pm_fps<<<4, 256, 0, stream>>>(posin, IDX, n, m);
    pm_knn<<<cdiv((long)4*m,128), 128, 0, stream>>>(posin, IDX, NBR, n, m);
    pm_gather_pos<<<cdiv((long)4*m*3,256), 256, 0, stream>>>(posin, IDX, posout, n, m, (long)4*m*3);

    long sigtot = (long)4*m*KNN_K*di;
    pm_sigma_partial<<<1024, 256, 0, stream>>>(xin, IDX, NBR, SIGP, n, m, di, sigtot);
    pm_reduce1<<<1, 256, 0, stream>>>(SIGP, 1024, SIG);
    long xtot = (long)4*n*di;
    float denom = 1.0f / ((float)KNN_K * (float)(4*n) * (float)di);
    pm_scale<<<cdiv(xtot,256), 256, 0, stream>>>(xin, cp[i].alpha, cp[i].beta, SIG, denom, xtot, di);

    // transfer: t = relu(bn(x @ tW))
    const int M = 4*n; const long Mt = (long)M*dk;
    gemm(xin, wtT[i], CB, M, dk, di);
    bnstats(CB, M, dk);
    pm_bn_relu<<<cdiv(Mt,256), 256, 0, stream>>>(CB, MEAN, RSTD, cp[i].tg, cp[i].tbb, ACTA, Mt, dk);
    // pre residual block on all n nodes: out = bn(relu(bn(t@W1))@W2) + t
    gemm(ACTA, wtP1[i], CB, M, dk, dk);
    bnstats(CB, M, dk);
    pm_bn_relu<<<cdiv(Mt,256), 256, 0, stream>>>(CB, MEAN, RSTD, cp[i].pg1, cp[i].pb1, ACTB, Mt, dk);
    gemm(ACTB, wtP2[i], CB, M, dk, dk);
    bnstats(CB, M, dk);
    pm_bn_add<<<cdiv(Mt,256), 256, 0, stream>>>(CB, MEAN, RSTD, cp[i].pg2, cp[i].pb2, ACTA, ACTB, Mt, dk);

    // max aggregation over neighbors -> [4*m, dk]
    long At = (long)4*m*dk;
    pm_gather_max<<<cdiv(At,256), 256, 0, stream>>>(ACTB, NBR, AGG, m, n, dk, At);

    // pos residual block on anchors
    const int Mm = 4*m;
    gemm(AGG, wtQ1[i], CB, Mm, dk, dk);
    bnstats(CB, Mm, dk);
    pm_bn_relu<<<cdiv(At,256), 256, 0, stream>>>(CB, MEAN, RSTD, cp[i].qg1, cp[i].qb1, ACTA, At, dk);
    gemm(ACTA, wtQ2[i], CB, Mm, dk, dk);
    bnstats(CB, Mm, dk);
    pm_bn_add<<<cdiv(At,256), 256, 0, stream>>>(CB, MEAN, RSTD, cp[i].qg2, cp[i].qb2, AGG, xout, At, dk);

    xin = xout; posin = posout;
  }

  // ---- global max pool + classifier ----
  pm_global_max<<<cdiv((long)4*512,256), 256, 0, stream>>>(xin, GP, 512, 512);
  pm_classifier<<<1, 256, 0, stream>>>(GP, cW[0], cg[0], cbb[0], CT, 512, 256);
  pm_classifier<<<1, 128, 0, stream>>>(CT, cW[1], cg[1], cbb[1], (float*)d_out, 256, 128);
}
